// Attention_11252814315826
// MI455X (gfx1250) — matrix operand order repaired and checked by reading, not compiled, not run on hardware
//
#include <hip/hip_runtime.h>
#include <stdint.h>

typedef __attribute__((ext_vector_type(16))) _Float16 v16h;
typedef __attribute__((ext_vector_type(8)))  _Float16 v8h;
typedef __attribute__((ext_vector_type(2)))  _Float16 h2;
typedef __attribute__((ext_vector_type(8)))  float    v8f;
typedef __attribute__((ext_vector_type(4)))  int      v4i;

#define GAS __attribute__((address_space(1)))
#define LAS __attribute__((address_space(3)))

#if defined(__has_builtin) && __has_builtin(__builtin_amdgcn_global_load_async_to_lds_b128)
#define HAVE_ASYNC_LDS 1
#else
#define HAVE_ASYNC_LDS 0
#endif

#if defined(__has_builtin) && __has_builtin(__builtin_amdgcn_s_wait_asynccnt)
#define WAIT_ASYNC(n) __builtin_amdgcn_s_wait_asynccnt(n)
#else
#define WAIT_ASYNC(n) asm volatile("s_wait_asynccnt %0" :: "n"(n) : "memory")
#endif

namespace {
constexpr int B  = 2;
constexpr int H  = 16;
constexpr int S  = 2048;
constexpr int D  = 64;
constexpr int BM = 128;   // query rows per block (8 waves x 16)
constexpr int BN = 64;    // kv tile
constexpr float NEG_BIG = -1.0e12f;
}

__device__ __forceinline__ v16h load_frag16(const _Float16* lo, const _Float16* hi) {
    v8h a = *(const v8h*)lo;      // ds_load_b128
    v8h b = *(const v8h*)hi;      // ds_load_b128
    return __builtin_shufflevector(a, b, 0,1,2,3,4,5,6,7,8,9,10,11,12,13,14,15);
}

__global__ __launch_bounds__(256)
void attn_flash_wmma(const float* __restrict__ Q, const float* __restrict__ K,
                     const float* __restrict__ V, const int* __restrict__ mask,
                     float* __restrict__ O)
{
    __shared__ float    Kraw[2][D][BN];        // async landing pad, [d][t] 32 KB
    __shared__ float    Vraw[2][BN][D];        // async landing pad, [t][d] 32 KB
    __shared__ _Float16 K16[BN][D];            // f16 K^T tile: [t][d]   8 KB
    __shared__ _Float16 V16[D][BN];            // f16 V^T tile: [d][t]   8 KB
    __shared__ _Float16 Psh[8][16][BN];        // per-wave P scratch    16 KB

    const int tid  = threadIdx.x;
    const int lane = tid & 31;
    const int wave = tid >> 5;
    const int half = lane >> 4;      // wave32: lane group 0..15 / 16..31
    const int l16  = lane & 15;

    const int qtiles = S / BM;                 // 16
    const int qt = blockIdx.x % qtiles;
    const int bh = blockIdx.x / qtiles;        // 0..31
    const int b  = bh / H;

    const float scale = 0.125f;                // 1/sqrt(64)
    const float* Qb = Q + (size_t)bh * S * D;
    const float* Kb = K + (size_t)bh * D * S;  // key pre-transposed: [D][S]
    const float* Vb = V + (size_t)bh * S * D;
    float*       Ob = O + (size_t)bh * S * D;

    const int q0w = qt * BM + wave * 16;       // this wave's first query row

    // ---- Q A-fragments (16x64 -> two K=32 f16 frags), scale folded in ----
    // A 16x32 f16 layout: lane = M (l16), element j -> K = (j&7) + 16*(j>>3) + 8*half
    v16h qa0, qa1;
    {
        const float* qrow = Qb + (size_t)(q0w + l16) * D;
        #pragma unroll
        for (int j = 0; j < 16; ++j) {
            int d = (j & 7) + 16 * (j >> 3) + 8 * half;
            qa0[j] = (_Float16)(qrow[d]      * scale);
            qa1[j] = (_Float16)(qrow[d + 32] * scale);
        }
    }

    // additive query-row mask: C/D layout row M = r + 8*half
    float addm[8];
    #pragma unroll
    for (int r = 0; r < 8; ++r)
        addm[r] = mask[b * S + q0w + r + 8 * half] ? 0.0f : NEG_BIG;

    v8f acc[4] = {v8f{}, v8f{}, v8f{}, v8f{}};   // O accum, 16x64: acc[n][r]
    float mrow[8], lrow[8];
    #pragma unroll
    for (int r = 0; r < 8; ++r) { mrow[r] = -3.0e38f; lrow[r] = 0.0f; }

    // ---- raw KV tile staging (async DMA to LDS, fp32) ----
    auto stage = [&](int buf, int t0) {
        #pragma unroll
        for (int c = 0; c < 4; ++c) {
            int linear = c * 256 + tid;          // 0..1023
            int row = linear >> 4;               // 0..63
            int col = (linear & 15) << 2;        // float4 column
            const float* gk = Kb + (size_t)row * S + t0 + col;   // K[d][t0+col..]
            const float* gv = Vb + (size_t)(t0 + row) * D + col; // V[t0+row][col..]
#if HAVE_ASYNC_LDS
            __builtin_amdgcn_global_load_async_to_lds_b128(
                (GAS v4i*)(uintptr_t)gk,
                (LAS v4i*)(uint32_t)(uintptr_t)&Kraw[buf][row][col], 0, 0);
            __builtin_amdgcn_global_load_async_to_lds_b128(
                (GAS v4i*)(uintptr_t)gv,
                (LAS v4i*)(uint32_t)(uintptr_t)&Vraw[buf][row][col], 0, 0);
#else
            *(float4*)&Kraw[buf][row][col] = *(const float4*)gk;
            *(float4*)&Vraw[buf][row][col] = *(const float4*)gv;
#endif
        }
    };

    stage(0, 0);
    const int nt = S / BN;                       // 32 kv tiles
    for (int it = 0; it < nt; ++it) {
        const int buf = it & 1;
        if (it + 1 < nt) stage(buf ^ 1, (it + 1) * BN);
#if HAVE_ASYNC_LDS
        if (it + 1 < nt) { WAIT_ASYNC(8); } else { WAIT_ASYNC(0); }
#endif
        __syncthreads();   // raw tile ready; previous compute done with f16 tiles

        // ---- one-time cooperative fp32 -> f16 transposed conversion ----
        // Kraw[d][t] -> K16[t][d] ; Vraw[t][d] -> V16[d][t] (packed pair stores)
        #pragma unroll
        for (int c = 0; c < 8; ++c) {
            int p  = c * 256 + tid;              // 0..2047 pair index
            int i0 = p & 63;
            int j0 = (p >> 6) << 1;              // even index of the pair
            // K: pair along d at fixed t
            h2 kp = { (_Float16)Kraw[buf][j0][i0], (_Float16)Kraw[buf][j0 + 1][i0] };
            *(h2*)&K16[i0][j0] = kp;             // v_cvt_pk_f16_f32 + ds_store_b32
            // V: pair along t at fixed d
            h2 vp = { (_Float16)Vraw[buf][j0][i0], (_Float16)Vraw[buf][j0 + 1][i0] };
            *(h2*)&V16[i0][j0] = vp;
        }
        __syncthreads();   // f16 tiles ready

        // ---- S = Q * K^T tile (16 x 64), fp32 accum ----
        v8f sc[4];
        #pragma unroll
        for (int n = 0; n < 4; ++n) {
            // B 32x16 f16 layout: lane = N (l16), element j -> K(d) = (j&7) + 16*(j>>3) + 8*half
            v16h kb0 = load_frag16(&K16[n * 16 + l16][8 * half],
                                   &K16[n * 16 + l16][16 + 8 * half]);
            v16h kb1 = load_frag16(&K16[n * 16 + l16][32 + 8 * half],
                                   &K16[n * 16 + l16][48 + 8 * half]);
            v8f c = {};
            c = __builtin_amdgcn_wmma_f32_16x16x32_f16(false, qa0, false, kb0,
                                                       (short)0, c, false, false);
            c = __builtin_amdgcn_wmma_f32_16x16x32_f16(false, qa1, false, kb1,
                                                       (short)0, c, false, false);
            sc[n] = c;
        }

        // ---- online softmax over this 64-key tile ----
        #pragma unroll
        for (int r = 0; r < 8; ++r) {
            float t = sc[0][r] + addm[r];
            sc[0][r] = t;
            #pragma unroll
            for (int n = 1; n < 4; ++n) {
                float v = sc[n][r] + addm[r];
                sc[n][r] = v;
                t = fmaxf(t, v);
            }
            // row is striped across the 16-lane group (same half) -> reduce
            #pragma unroll
            for (int off = 1; off < 16; off <<= 1)
                t = fmaxf(t, __shfl_xor(t, off, 32));

            float mnew  = fmaxf(mrow[r], t);
            float alpha = __expf(mrow[r] - mnew);
            mrow[r] = mnew;

            float ps = 0.0f;
            #pragma unroll
            for (int n = 0; n < 4; ++n) {
                float p = __expf(sc[n][r] - mnew);
                sc[n][r] = p;
                ps += p;
            }
            #pragma unroll
            for (int off = 1; off < 16; off <<= 1)
                ps += __shfl_xor(ps, off, 32);
            lrow[r] = lrow[r] * alpha + ps;

            #pragma unroll
            for (int n = 0; n < 4; ++n)
                acc[n][r] *= alpha;
        }

        // ---- relayout P: C-layout -> A-fragment via per-wave LDS scratch ----
        #pragma unroll
        for (int n = 0; n < 4; ++n)
            #pragma unroll
            for (int r = 0; r < 8; ++r)
                Psh[wave][r + 8 * half][n * 16 + l16] = (_Float16)sc[n][r];
        // same-wave DS ops complete in order (DScnt in-order), no barrier needed

        // A frag element j -> t = (j&7) + 16*(j>>3) + 8*half : two 16B runs
        v16h pa0 = load_frag16(&Psh[wave][l16][8 * half],
                               &Psh[wave][l16][16 + 8 * half]);
        v16h pa1 = load_frag16(&Psh[wave][l16][32 + 8 * half],
                               &Psh[wave][l16][48 + 8 * half]);

        // ---- O += P * V ----
        #pragma unroll
        for (int n = 0; n < 4; ++n) {
            // B frag: element j -> K(t) = (j&7) + 16*(j>>3) + 8*half at fixed d = n*16+l16
            v16h vb0 = load_frag16(&V16[n * 16 + l16][8 * half],
                                   &V16[n * 16 + l16][16 + 8 * half]);
            v16h vb1 = load_frag16(&V16[n * 16 + l16][32 + 8 * half],
                                   &V16[n * 16 + l16][48 + 8 * half]);
            acc[n] = __builtin_amdgcn_wmma_f32_16x16x32_f16(false, pa0, false, vb0,
                                                            (short)0, acc[n], false, false);
            acc[n] = __builtin_amdgcn_wmma_f32_16x16x32_f16(false, pa1, false, vb1,
                                                            (short)0, acc[n], false, false);
        }

        __syncthreads();   // all reads of f16 tiles done before next convert
    }

    // ---- normalize and write O (C/D layout: M = r+8*half, N = n*16+l16) ----
    #pragma unroll
    for (int r = 0; r < 8; ++r) {
        float inv = 1.0f / lrow[r];
        int row = q0w + r + 8 * half;
        #pragma unroll
        for (int n = 0; n < 4; ++n)
            Ob[(size_t)row * D + n * 16 + l16] = acc[n][r] * inv;
    }
}

extern "C" void kernel_launch(void* const* d_in, const int* in_sizes, int n_in,
                              void* d_out, int out_size, void* d_ws, size_t ws_size,
                              hipStream_t stream) {
    (void)in_sizes; (void)n_in; (void)out_size; (void)d_ws; (void)ws_size;
    const float* Q    = (const float*)d_in[0];
    const float* K    = (const float*)d_in[1];   // [B,H,D,S] pre-transposed
    const float* V    = (const float*)d_in[2];
    const int*   mask = (const int*)d_in[3];
    float*       O    = (float*)d_out;

    dim3 grid(B * H * (S / BM));   // 32 * 16 = 512 blocks
    dim3 block(256);               // 8 waves (wave32)
    attn_flash_wmma<<<grid, block, 0, stream>>>(Q, K, V, mask, O);
}
